// BidirectionAttBlock_46789373722935
// MI455X (gfx1250) — matrix operand order repaired and checked by reading, not compiled, not run on hardware
//
#include <hip/hip_runtime.h>
#include <hip/hip_bf16.h>
#include <math.h>

// ---------------------------------------------------------------------------
// BidirectionAttBlock for MI455X (gfx1250), wave32 + WMMA f16 (f32 accum).
// B=128, ND=256, NP=1024, L=128, NB=4.
// ---------------------------------------------------------------------------

#define B_  128
#define ND_ 256
#define NP2 1024
#define L_  128
#define NB_ 4

typedef _Float16 v16h __attribute__((ext_vector_type(16)));
typedef _Float16 v8h  __attribute__((ext_vector_type(8)));
typedef float    v8f  __attribute__((ext_vector_type(8)));

__device__ __forceinline__ v8f wmma_f16(v16h a, v16h b, v8f c) {
  // D = A(16x32 f16) * B(32x16 f16) + C(16x16 f32)
  return __builtin_amdgcn_wmma_f32_16x16x32_f16(false, a, false, b, (short)0, c,
                                                false, false);
}

// Branchless tanh: CDNA5 hardware V_TANH_F32 (probe-confirmed to lower).
__device__ __forceinline__ float fast_tanh(float x) {
#if __has_builtin(__builtin_amdgcn_tanhf)
  return __builtin_amdgcn_tanhf(x);
#else
  float xc = fminf(fmaxf(x, -9.0f), 9.0f);
  float e = __expf(xc * 2.0f);
  return 1.0f - 2.0f * __builtin_amdgcn_rcpf(e + 1.0f);
#endif
}

// A fragment (16x32, MxK) from row-major f16 [row][k], row stride ld halves.
// lane: m = lane&15, hi = lane>>4. idx 0..7 -> k = hi*8+idx ;
// idx 8..15 -> k = 16 + hi*8 + (idx-8).   (ISA 7.12.2)
__device__ __forceinline__ v16h load_afrag(const _Float16* base, int ld,
                                           int lane, int k0) {
  int m = lane & 15, hi = lane >> 4;
  const _Float16* p0 = base + m * ld + k0 + hi * 8;
  v8h lo = *(const v8h*)(p0);
  v8h hs = *(const v8h*)(p0 + 16);
  v16h r;
#pragma unroll
  for (int e = 0; e < 8; ++e) { r[e] = lo[e]; r[8 + e] = hs[e]; }
  return r;
}

// B fragment (32x16, KxN) where B is stored as [n][k] row-major (k contiguous),
// row stride ld halves. idx 0..7 -> k = hi*8+idx ; idx 8..15 -> k = 16 + hi*8 + (idx-8).
__device__ __forceinline__ v16h load_bfrag(const _Float16* base, int ld,
                                           int lane, int k0) {
  int n = lane & 15, hi = lane >> 4;
  const _Float16* p0 = base + n * ld + k0 + hi * 8;
  v8h lo = *(const v8h*)(p0);
  v8h hs = *(const v8h*)(p0 + 16);
  v16h r;
#pragma unroll
  for (int e = 0; e < 8; ++e) { r[e] = lo[e]; r[8 + e] = hs[e]; }
  return r;
}

// ---------------------------------------------------------------------------
// Kernel 1a: f32 -> f16 convert
__global__ void __launch_bounds__(256) k_convert(const float* __restrict__ s,
                                                 _Float16* __restrict__ d, int n) {
  int i = blockIdx.x * 256 + threadIdx.x;
  if (i < n) d[i] = (_Float16)s[i];
}

// Kernel 1b: transpose [4,L,L] f32 -> [4,L,L] f16 with Wt[i][m][l] = W[i][l][m]
__global__ void __launch_bounds__(256) k_transpose(const float* __restrict__ W,
                                                   _Float16* __restrict__ Wt) {
  int idx = blockIdx.x * 256 + threadIdx.x;          // 4*128*128 = 65536
  int i = idx >> 14, rem = idx & 16383, m = rem >> 7, l = rem & 127;
  Wt[idx] = (_Float16)W[(i << 14) + (l << 7) + m];
}

// ---------------------------------------------------------------------------
// Kernel 2: out[b,i,row] = tanh(X[b]@W[i] + bias[i]) . wvec[i, woff:woff+L]
// X: [B, rows, L] f16 row-major; Wt: [4, L, L] f16 (Wt[i][m][l] = W[i][l][m]).
// One wave per 16-row tile, 8 waves per block.
__global__ void __launch_bounds__(256) k_rowdot(const _Float16* __restrict__ X,
                                                const _Float16* __restrict__ Wt,
                                                const float* __restrict__ bias,
                                                const float* __restrict__ wvec,
                                                int woff,
                                                float* __restrict__ out,
                                                int rows) {
  int wave = threadIdx.x >> 5, lane = threadIdx.x & 31;
  int hi = lane >> 4, nc = lane & 15;
  int tilesPerBI = rows >> 4;
  int gwave = blockIdx.x * 8 + wave;
  int tile = gwave % tilesPerBI;
  int bi = gwave / tilesPerBI;            // b*4 + i
  int i = bi & 3;
  const _Float16* Xrow = X + ((long)(bi >> 2) * rows + tile * 16) * L_;
  const _Float16* Wti = Wt + (i << 14);
  const float* bb = bias + i * L_;
  const float* w = wvec + i * 2 * L_ + woff;

  v16h afr[4];
#pragma unroll
  for (int kt = 0; kt < 4; ++kt) afr[kt] = load_afrag(Xrow, L_, lane, kt * 32);

  float part[8];
#pragma unroll
  for (int v = 0; v < 8; ++v) part[v] = 0.f;

  for (int nt = 0; nt < 8; ++nt) {
    // batch the 4 B fragments so loads issue as one clause
    v16h bf[4];
#pragma unroll
    for (int kt = 0; kt < 4; ++kt)
      bf[kt] = load_bfrag(Wti + nt * 16 * L_, L_, lane, kt * 32);
    v8f c = {0.f, 0.f, 0.f, 0.f, 0.f, 0.f, 0.f, 0.f};
#pragma unroll
    for (int kt = 0; kt < 4; ++kt) c = wmma_f16(afr[kt], bf[kt], c);

    int ncol = nt * 16 + nc;
    float bv = bb[ncol], wv = w[ncol];
#pragma unroll
    for (int v = 0; v < 8; ++v) part[v] += fast_tanh(c[v] + bv) * wv;
  }
  // reduce the 16 feature-columns held by lanes within each half
#pragma unroll
  for (int v = 0; v < 8; ++v) {
    float r = part[v];
    r += __shfl_xor(r, 1); r += __shfl_xor(r, 2);
    r += __shfl_xor(r, 4); r += __shfl_xor(r, 8);
    part[v] = r;
  }
  if (nc == 0) {
    float* o = out + (long)bi * rows + tile * 16 + hi * 8;
#pragma unroll
    for (int v = 0; v < 8; ++v) o[v] = part[v];
  }
}

// ---------------------------------------------------------------------------
// Kernel 3: per (b,i):  DU = drug@U[i]  (LDS, f16), then
//   th[n,p]   = tanh(DU[n,:].prot[p,:])
//   rowsum_c[n] = dm[n] * sum_p th*(pm[p]*s[p]) ; colsum_p[p] = pm[p]*sum_n th*(dm[n]*t[n])
#define DU_LD 136
__global__ void __launch_bounds__(256) k_interact(
    const _Float16* __restrict__ drug_h, const _Float16* __restrict__ prot_h,
    const _Float16* __restrict__ Ut,
    const float* __restrict__ drug_mask, const float* __restrict__ prot_mask,
    const float* __restrict__ s_vec, const float* __restrict__ t_vec,
    float* __restrict__ rowsum_c, float* __restrict__ colsum_p) {
  __shared__ _Float16 du[ND_ * DU_LD];
  __shared__ float ps_l[NP2], pm[NP2], dt_l[ND_], dm[ND_];
  __shared__ float partial[8][ND_];

  int b = blockIdx.x >> 2, i = blockIdx.x & 3;
  int bi = (b << 2) + i;
  int tid = threadIdx.x, wave = tid >> 5, lane = tid & 31;
  int hi = lane >> 4, lx = lane & 15;

  for (int k = tid; k < NP2; k += 256) {
    float pmv = prot_mask[((long)b << 10) + k];
    pm[k] = pmv;
    ps_l[k] = pmv * s_vec[((long)bi << 10) + k];   // pm[p]*s[p]
  }
  {
    float dmv = drug_mask[((long)b << 8) + tid];
    dm[tid] = dmv;
    dt_l[tid] = dmv * t_vec[((long)bi << 8) + tid]; // dm[n]*t[n]
  }

  // ---- phase 1: DU[n][m] = sum_l drug[n,l]*U[l,m], f16 into LDS
  const _Float16* dbase = drug_h + ((long)b << 15);   // 256*128
  const _Float16* uti = Ut + (i << 14);
  for (int nt = wave * 2; nt < wave * 2 + 2; ++nt) {
    v16h afr[4];
#pragma unroll
    for (int kt = 0; kt < 4; ++kt)
      afr[kt] = load_afrag(dbase + nt * 16 * L_, L_, lane, kt * 32);
    for (int mt = 0; mt < 8; ++mt) {
      v16h bf[4];
#pragma unroll
      for (int kt = 0; kt < 4; ++kt)
        bf[kt] = load_bfrag(uti + mt * 16 * L_, L_, lane, kt * 32);
      v8f c = {0.f, 0.f, 0.f, 0.f, 0.f, 0.f, 0.f, 0.f};
#pragma unroll
      for (int kt = 0; kt < 4; ++kt) c = wmma_f16(afr[kt], bf[kt], c);

      int row = nt * 16 + hi * 8, col = mt * 16 + lx;
#pragma unroll
      for (int v = 0; v < 8; ++v) du[(row + v) * DU_LD + col] = (_Float16)c[v];
    }
  }
  __syncthreads();

  // ---- phase 2: A tiles, fused row/col weighted sums
  const _Float16* pbase = prot_h + ((long)b << 17);   // 1024*128
  float colacc[8];
#pragma unroll
  for (int pi = 0; pi < 8; ++pi) colacc[pi] = 0.f;

  for (int nt = 0; nt < 16; ++nt) {
    v16h afr[4];
#pragma unroll
    for (int kt = 0; kt < 4; ++kt)
      afr[kt] = load_afrag(du + nt * 16 * DU_LD, DU_LD, lane, kt * 32);
    float rowacc[8];
#pragma unroll
    for (int v = 0; v < 8; ++v) rowacc[v] = 0.f;

    for (int pi = 0; pi < 8; ++pi) {
      int pt = pi * 8 + wave;                      // wave-private p-tiles
      v16h bf[4];
#pragma unroll
      for (int kt = 0; kt < 4; ++kt)
        bf[kt] = load_bfrag(pbase + pt * 16 * L_, L_, lane, kt * 32);
      v8f c = {0.f, 0.f, 0.f, 0.f, 0.f, 0.f, 0.f, 0.f};
#pragma unroll
      for (int kt = 0; kt < 4; ++kt) c = wmma_f16(afr[kt], bf[kt], c);

      int p = pt * 16 + lx;
      float psv = ps_l[p];
#pragma unroll
      for (int v = 0; v < 8; ++v) {
        int n = nt * 16 + hi * 8 + v;
        float th = fast_tanh(c[v]);
        rowacc[v] += th * psv;       // * dm[n] applied at final write
        colacc[pi] += th * dt_l[n];  // * pm[p] applied at final write
      }
    }
    // reduce rowacc over the 16 p-columns in each lane half
#pragma unroll
    for (int v = 0; v < 8; ++v) {
      float r = rowacc[v];
      r += __shfl_xor(r, 1); r += __shfl_xor(r, 2);
      r += __shfl_xor(r, 4); r += __shfl_xor(r, 8);
      if (lx == 0) partial[wave][nt * 16 + hi * 8 + v] = r;
    }
  }
  // colsum: lane l and l+16 hold same p (different n halves)
#pragma unroll
  for (int pi = 0; pi < 8; ++pi) {
    float cc = colacc[pi] + __shfl_xor(colacc[pi], 16);
    if (hi == 0) {
      int p = (pi * 8 + wave) * 16 + lx;
      colsum_p[((long)bi << 10) + p] = cc * pm[p];
    }
  }
  __syncthreads();
  // deterministic rowsum: sum 8 wave partials per n, apply drug mask
  {
    float s = 0.f;
#pragma unroll
    for (int w = 0; w < 8; ++w) s += partial[w][tid];
    rowsum_c[((long)bi << 8) + tid] = s * dm[tid];
  }
}

// ---------------------------------------------------------------------------
// Kernel 4: masked softmax + attention pooling, per (b,i)
__global__ void __launch_bounds__(256) k_attpool(
    const float* __restrict__ drug_mat, const float* __restrict__ prot_mat,
    const float* __restrict__ drug_mask, const float* __restrict__ prot_mask,
    const float* __restrict__ u_c, const float* __restrict__ rowsum_c,
    const float* __restrict__ att_c_b,
    const float* __restrict__ u_p, const float* __restrict__ colsum_p,
    const float* __restrict__ att_p_b,
    float* __restrict__ cf, float* __restrict__ pf) {
  __shared__ float wgt[NP2];
  __shared__ float red[8];
  int b = blockIdx.x >> 2, i = blockIdx.x & 3;
  int bi = (b << 2) + i;
  int tid = threadIdx.x, wave = tid >> 5, lane = tid & 31;

  // ---------- drug side (256 logits) ----------
  float v = u_c[(long)bi * ND_ + tid] + rowsum_c[(long)bi * ND_ + tid] + att_c_b[i];
  float m = v;
  m = fmaxf(m, __shfl_xor(m, 1));  m = fmaxf(m, __shfl_xor(m, 2));
  m = fmaxf(m, __shfl_xor(m, 4));  m = fmaxf(m, __shfl_xor(m, 8));
  m = fmaxf(m, __shfl_xor(m, 16));
  if (lane == 0) red[wave] = m;
  __syncthreads();
  m = red[0];
#pragma unroll
  for (int w = 1; w < 8; ++w) m = fmaxf(m, red[w]);
  __syncthreads();
  float e = __expf(v - m) * drug_mask[(long)b * ND_ + tid];
  float s = e;
  s += __shfl_xor(s, 1); s += __shfl_xor(s, 2); s += __shfl_xor(s, 4);
  s += __shfl_xor(s, 8); s += __shfl_xor(s, 16);
  if (lane == 0) red[wave] = s;
  __syncthreads();
  s = 0.f;
#pragma unroll
  for (int w = 0; w < 8; ++w) s += red[w];
  __syncthreads();
  wgt[tid] = e / (s + 1e-6f);
  __syncthreads();
  if (tid < L_) {
    float acc = 0.f;
    const float* dp = drug_mat + (long)b * ND_ * L_ + tid;
    for (int n = 0; n < ND_; ++n) acc += wgt[n] * dp[(long)n * L_];
    cf[(long)bi * L_ + tid] = acc;
  }
  __syncthreads();

  // ---------- protein side (1024 logits, 4 per thread) ----------
  float vv[4], lm = -INFINITY;
#pragma unroll
  for (int q = 0; q < 4; ++q) {
    int n = q * 256 + tid;
    vv[q] = u_p[(long)bi * NP2 + n] + colsum_p[(long)bi * NP2 + n] + att_p_b[i];
    lm = fmaxf(lm, vv[q]);
  }
  lm = fmaxf(lm, __shfl_xor(lm, 1));  lm = fmaxf(lm, __shfl_xor(lm, 2));
  lm = fmaxf(lm, __shfl_xor(lm, 4));  lm = fmaxf(lm, __shfl_xor(lm, 8));
  lm = fmaxf(lm, __shfl_xor(lm, 16));
  if (lane == 0) red[wave] = lm;
  __syncthreads();
  lm = red[0];
#pragma unroll
  for (int w = 1; w < 8; ++w) lm = fmaxf(lm, red[w]);
  __syncthreads();
  float ev[4], ls = 0.f;
#pragma unroll
  for (int q = 0; q < 4; ++q) {
    int n = q * 256 + tid;
    ev[q] = __expf(vv[q] - lm) * prot_mask[(long)b * NP2 + n];
    ls += ev[q];
  }
  ls += __shfl_xor(ls, 1); ls += __shfl_xor(ls, 2); ls += __shfl_xor(ls, 4);
  ls += __shfl_xor(ls, 8); ls += __shfl_xor(ls, 16);
  if (lane == 0) red[wave] = ls;
  __syncthreads();
  ls = 0.f;
#pragma unroll
  for (int w = 0; w < 8; ++w) ls += red[w];
  __syncthreads();
#pragma unroll
  for (int q = 0; q < 4; ++q) wgt[q * 256 + tid] = ev[q] / (ls + 1e-6f);
  __syncthreads();
  if (tid < L_) {
    float acc = 0.f;
    const float* pp = prot_mat + (long)b * NP2 * L_ + tid;
    for (int n = 0; n < NP2; ++n) acc += wgt[n] * pp[(long)n * L_];
    pf[(long)bi * L_ + tid] = acc;
  }
}

// ---------------------------------------------------------------------------
// Kernel 5: cf_fin = cf_all@comb_c_w + b ; pf_fin likewise ;
//           out[b, r*L + j] = leaky_relu(cf_fin[r]*pf_fin[j], 0.1)
__global__ void __launch_bounds__(128) k_final(
    const float* __restrict__ cf, const float* __restrict__ pf,
    const float* __restrict__ comb_c_w, const float* __restrict__ comb_c_b,
    const float* __restrict__ comb_p_w, const float* __restrict__ comb_p_b,
    float* __restrict__ out) {
  __shared__ float cff[L_], pff[L_];
  int b = blockIdx.x, j = threadIdx.x;
  float a = comb_c_b[j], q = comb_p_b[j];
  const float* cfb = cf + (long)b * (NB_ * L_);
  const float* pfb = pf + (long)b * (NB_ * L_);
  for (int k = 0; k < NB_ * L_; ++k) {
    a += cfb[k] * comb_c_w[(long)k * L_ + j];
    q += pfb[k] * comb_p_w[(long)k * L_ + j];
  }
  cff[j] = a; pff[j] = q;
  __syncthreads();
  float* o = out + (long)b * (L_ * L_);
  for (int r = 0; r < L_; ++r) {
    float v = cff[r] * pff[j];
    o[r * L_ + j] = (v > 0.f) ? v : 0.1f * v;
  }
}

// ---------------------------------------------------------------------------
extern "C" void kernel_launch(void* const* d_in, const int* in_sizes, int n_in,
                              void* d_out, int out_size, void* d_ws, size_t ws_size,
                              hipStream_t stream) {
  (void)in_sizes; (void)n_in; (void)out_size; (void)ws_size;
  const float* drug_mat  = (const float*)d_in[0];
  const float* drug_mask = (const float*)d_in[1];
  const float* prot_mat  = (const float*)d_in[2];
  const float* prot_mask = (const float*)d_in[3];
  const float* U         = (const float*)d_in[4];
  const float* t_c2p_w   = (const float*)d_in[5];
  const float* t_c2p_b   = (const float*)d_in[6];
  const float* t_p2c_w   = (const float*)d_in[7];
  const float* t_p2c_b   = (const float*)d_in[8];
  const float* bh_c_w    = (const float*)d_in[9];
  const float* bh_c_b    = (const float*)d_in[10];
  const float* bh_p_w    = (const float*)d_in[11];
  const float* bh_p_b    = (const float*)d_in[12];
  const float* att_c_w   = (const float*)d_in[13];
  const float* att_c_b   = (const float*)d_in[14];
  const float* att_p_w   = (const float*)d_in[15];
  const float* att_p_b   = (const float*)d_in[16];
  const float* comb_c_w  = (const float*)d_in[17];
  const float* comb_c_b  = (const float*)d_in[18];
  const float* comb_p_w  = (const float*)d_in[19];
  const float* comb_p_b  = (const float*)d_in[20];
  float* out = (float*)d_out;

  // workspace carve-up (bytes; all 256B aligned). Total ~51 MB.
  char* ws = (char*)d_ws;
  size_t off = 0;
  auto carve = [&](size_t bytes) { char* p = ws + off; off += (bytes + 255) & ~(size_t)255; return p; };
  _Float16* drug_h  = (_Float16*)carve((size_t)B_ * ND_ * L_ * 2);
  _Float16* prot_h  = (_Float16*)carve((size_t)B_ * NP2 * L_ * 2);
  _Float16* Ut_h    = (_Float16*)carve((size_t)NB_ * L_ * L_ * 2);
  _Float16* Wt_c2p  = (_Float16*)carve((size_t)NB_ * L_ * L_ * 2);
  _Float16* Wt_p2c  = (_Float16*)carve((size_t)NB_ * L_ * L_ * 2);
  _Float16* Wt_bhc  = (_Float16*)carve((size_t)NB_ * L_ * L_ * 2);
  _Float16* Wt_bhp  = (_Float16*)carve((size_t)NB_ * L_ * L_ * 2);
  float* t_vec    = (float*)carve((size_t)B_ * NB_ * ND_ * 4);
  float* u_c      = (float*)carve((size_t)B_ * NB_ * ND_ * 4);
  float* s_vec    = (float*)carve((size_t)B_ * NB_ * NP2 * 4);
  float* u_p      = (float*)carve((size_t)B_ * NB_ * NP2 * 4);
  float* rowsum_c = (float*)carve((size_t)B_ * NB_ * ND_ * 4);
  float* colsum_p = (float*)carve((size_t)B_ * NB_ * NP2 * 4);
  float* cf       = (float*)carve((size_t)B_ * NB_ * L_ * 4);
  float* pf       = (float*)carve((size_t)B_ * NB_ * L_ * 4);

  // 1) precision conversions + weight transposes
  {
    int n = B_ * ND_ * L_;
    k_convert<<<n / 256, 256, 0, stream>>>(drug_mat, drug_h, n);
  }
  {
    int n = B_ * NP2 * L_;
    k_convert<<<n / 256, 256, 0, stream>>>(prot_mat, prot_h, n);
  }
  k_transpose<<<256, 256, 0, stream>>>(U,       Ut_h);
  k_transpose<<<256, 256, 0, stream>>>(t_c2p_w, Wt_c2p);
  k_transpose<<<256, 256, 0, stream>>>(t_p2c_w, Wt_p2c);
  k_transpose<<<256, 256, 0, stream>>>(bh_c_w,  Wt_bhc);
  k_transpose<<<256, 256, 0, stream>>>(bh_p_w,  Wt_bhp);

  // 2) score-vector precomputations (WMMA GEMM + tanh + dot)
  //    t[n] = tanh(drug@t_c2p_w+b).att_p_w[L:2L] ; u_c[n] = tanh(drug@bh_c_w+b).att_c_w[0:L]
  //    s[p] = tanh(prot@t_p2c_w+b).att_c_w[L:2L] ; u_p[p] = tanh(prot@bh_p_w+b).att_p_w[0:L]
  k_rowdot<<<(B_ * NB_ * (ND_ / 16)) / 8, 256, 0, stream>>>(drug_h, Wt_c2p, t_c2p_b, att_p_w, L_, t_vec, ND_);
  k_rowdot<<<(B_ * NB_ * (ND_ / 16)) / 8, 256, 0, stream>>>(drug_h, Wt_bhc, bh_c_b, att_c_w, 0,  u_c,   ND_);
  k_rowdot<<<(B_ * NB_ * (NP2 / 16)) / 8, 256, 0, stream>>>(prot_h, Wt_p2c, t_p2c_b, att_c_w, L_, s_vec, NP2);
  k_rowdot<<<(B_ * NB_ * (NP2 / 16)) / 8, 256, 0, stream>>>(prot_h, Wt_bhp, bh_p_b, att_p_w, 0,  u_p,   NP2);

  // 3) fused interaction: A = tanh((drug@U).prot^T)*mask, row/col weighted sums
  k_interact<<<B_ * NB_, 256, 0, stream>>>(drug_h, prot_h, Ut_h,
                                           drug_mask, prot_mask,
                                           s_vec, t_vec, rowsum_c, colsum_p);

  // 4) masked softmax + attention pooling
  k_attpool<<<B_ * NB_, 256, 0, stream>>>(drug_mat, prot_mat, drug_mask, prot_mask,
                                          u_c, rowsum_c, att_c_b,
                                          u_p, colsum_p, att_p_b, cf, pf);

  // 5) combine + leaky-relu outer product
  k_final<<<B_, 128, 0, stream>>>(cf, pf, comb_c_w, comb_c_b, comb_p_w, comb_p_b, out);
}
